// SPHConvNet_14104672600387
// MI455X (gfx1250) — matrix operand order repaired and checked by reading, not compiled, not run on hardware
//
#include <hip/hip_runtime.h>
#include <hip/hip_bf16.h>

// ---------------------------------------------------------------------------
// SPHConv on MI455X (gfx1250): 3-kernel pipeline.
//  ws layout: [knn_idx int 8192*64][knn_d2 f32 8192*64][z f16 8192*1024] ~20MB
//  WMMA fragments are loaded as contiguous b128 runs by storing the B-side
//  matrices K-major (transposed), per the ISA 7.12.2 lane layouts.
//  Output GEMM stages its f16 activation chunks with the gfx1250 async
//  global->LDS path (ASYNCcnt) when the toolchain exposes the builtins.
// ---------------------------------------------------------------------------

typedef __attribute__((ext_vector_type(16))) _Float16 v16h;
typedef __attribute__((ext_vector_type(8)))  _Float16 v8h;
typedef __attribute__((ext_vector_type(8)))  float    v8f;
typedef __attribute__((ext_vector_type(4)))  int      v4i;

#ifndef __has_builtin
#define __has_builtin(x) 0
#endif
#if __has_builtin(__builtin_amdgcn_global_load_async_to_lds_b128) && \
    __has_builtin(__builtin_amdgcn_s_wait_asynccnt)
#define USE_ASYNC_LDS 1
#else
#define USE_ASYNC_LDS 0
#endif

#define BATCH   2
#define NPTS    4096
#define CH      64
#define PATCH   64
#define NRAD    4
#define NPOINTS (BATCH * NPTS)
#define FDIM    1024   // CH * NRAD * (L_MAX+1) = 64*16

__device__ __forceinline__ v16h cat8(v8h lo, v8h hi) {
  return __builtin_shufflevector(lo, hi, 0, 1, 2, 3, 4, 5, 6, 7,
                                         8, 9, 10, 11, 12, 13, 14, 15);
}

// ------------------------------ Kernel 1: KNN ------------------------------
__global__ __launch_bounds__(128) void knn_kernel(const float* __restrict__ xyz,
                                                  int*   __restrict__ knn_idx,
                                                  float* __restrict__ knn_d2) {
  __shared__ float dist[NPTS];
  __shared__ float rv[128];
  __shared__ int   ri[128];
  const int tid = threadIdx.x;
  const int pt  = blockIdx.x;          // b*N + v
  const int b   = pt / NPTS;
  const float qx = xyz[pt * 3 + 0];
  const float qy = xyz[pt * 3 + 1];
  const float qz = xyz[pt * 3 + 2];
  const float* base = xyz + (size_t)b * NPTS * 3;
  for (int j = tid; j < NPTS; j += 128) {
    float dx = base[j * 3 + 0] - qx;
    float dy = base[j * 3 + 1] - qy;
    float dz = base[j * 3 + 2] - qz;
    dist[j] = dx * dx + dy * dy + dz * dz;
  }
  __syncthreads();
  for (int it = 0; it < PATCH; ++it) {
    float best = 3.4e38f; int bi = 0;
    for (int j = tid; j < NPTS; j += 128) {
      float d = dist[j];
      if (d < best) { best = d; bi = j; }
    }
    rv[tid] = best; ri[tid] = bi;
    __syncthreads();
    for (int s = 64; s > 0; s >>= 1) {
      if (tid < s) {
        if (rv[tid + s] < rv[tid] ||
            (rv[tid + s] == rv[tid] && ri[tid + s] < ri[tid])) {
          rv[tid] = rv[tid + s]; ri[tid] = ri[tid + s];
        }
      }
      __syncthreads();
    }
    if (tid == 0) {
      knn_idx[(size_t)pt * PATCH + it] = ri[0];
      knn_d2 [(size_t)pt * PATCH + it] = rv[0];
      dist[ri[0]] = 3.4e38f;
    }
    __syncthreads();
  }
}

// ----------------------- Spherical harmonics (l<=3) ------------------------
__device__ __forceinline__ void sph_y(float x, float y, float z, float* Y) {
  const float x2 = x * x, y2 = y * y, z2 = z * z;
  Y[0]  = 0.28209479177387814f;                     // Y00
  Y[1]  = 0.4886025119029199f * y;                  // l=1
  Y[2]  = 0.4886025119029199f * z;
  Y[3]  = -0.4886025119029199f * x;
  Y[4]  = 1.0925484305920792f * x * y;              // l=2
  Y[5]  = 1.0925484305920792f * y * z;
  Y[6]  = 0.31539156525252005f * (2.0f * z2 - x2 - y2);
  Y[7]  = -1.0925484305920792f * z * x;
  Y[8]  = 0.5462742152960396f * (x2 - y2);
  Y[9]  = 0.5900435899266435f * (3.0f * x2 * y - y * y2);   // l=3
  Y[10] = 2.890611442640554f  * x * y * z;
  Y[11] = 0.4570457994644658f * (4.0f * y * z2 - x2 * y - y * y2);
  Y[12] = 0.3731763325901154f * (2.0f * z * z2 - 3.0f * x2 * z - 3.0f * y2 * z);
  Y[13] = -0.4570457994644658f * (4.0f * z2 * x - x * x2 - x * y2);
  Y[14] = 1.445305721320277f  * (x2 * z - y2 * z);
  Y[15] = -0.5900435899266435f * (x * x2 - 3.0f * x * y2);
}

// --------- Kernel 2: per-point 64x64x64 contraction via WMMA f16 -----------
// D[c, rn] = sum_p sigT[c,p] * kern[p,rn];  A = sigT (row-major over p),
// B stored transposed as kernT[rn][p] so B fragments are contiguous over p.
#define SIGT_S 72   // halves; 144B rows (16B aligned)
#define KERT_S 80   // halves; 160B rows (16B aligned)
__global__ __launch_bounds__(128) void point_conv_kernel(
    const float* __restrict__ xyz, const float* __restrict__ signal,
    const int* __restrict__ knn_idx, const float* __restrict__ knn_d2,
    _Float16* __restrict__ zbuf) {
  __shared__ _Float16 kernT[PATCH][KERT_S];  // [rn][p]
  __shared__ _Float16 sigT[CH][SIGT_S];      // [c][p]
  __shared__ float    ysq[CH][66];           // squared contraction result
  __shared__ int      nidx[PATCH];
  __shared__ float    red[PATCH];

  const int tid = threadIdx.x;
  const int pt  = blockIdx.x;
  const int b   = pt / NPTS;

  float Y[16];
  float rad[NRAD];

  // ---- Phase A: neighbor geometry (one neighbor per thread 0..63) ----
  if (tid < PATCH) {
    const int j = knn_idx[(size_t)pt * PATCH + tid];
    nidx[tid] = j;
    const float cx = xyz[pt * 3 + 0];
    const float cy = xyz[pt * 3 + 1];
    const float cz = xyz[pt * 3 + 2];
    const float* nb = xyz + ((size_t)b * NPTS + j) * 3;
    const float rx = nb[0] - cx, ry = nb[1] - cy, rz = nb[2] - cz;
    const float nrm = sqrtf(rx * rx + ry * ry + rz * rz);
    const float inv = 1.0f / fmaxf(nrm, 0.01f);
    sph_y(rx * inv, ry * inv, rz * inv, Y);
    const float d  = sqrtf(fmaxf(knn_d2[(size_t)pt * PATCH + tid], 1e-4f));
    const float sg = 0.5f / 3.0f;                 // sigma = radius/(NR-1)
    const float inv2s2 = 1.0f / (2.0f * sg * sg);
    #pragma unroll
    for (int r = 0; r < NRAD; ++r) {
      const float dd = d - (float)r * sg;
      rad[r] = __expf(-dd * dd * inv2s2);
    }
    red[tid] = Y[0] * rad[0];                     // y_w contribution
  }
  __syncthreads();
  // reduce y_w over 64 neighbors
  for (int s = 32; s > 0; s >>= 1) {
    if (tid < s) red[tid] += red[tid + s];
    __syncthreads();
  }
  // ---- write normalized conv kernel, transposed: kernT[r*16+n][p] ----
  if (tid < PATCH) {
    const float scale = 1.0f / (red[0] + 1e-6f);
    #pragma unroll
    for (int r = 0; r < NRAD; ++r) {
      const float rs = rad[r] * scale;
      #pragma unroll
      for (int n = 0; n < 16; ++n)
        kernT[r * 16 + n][tid] = (_Float16)(Y[n] * rs);
    }
  }
  // ---- gather signal patch -> sigT[c][p] (f16), coalesced over c ----
  for (int e = tid; e < PATCH * CH; e += 128) {
    const int p = e >> 6, c = e & 63;
    sigT[c][p] = (_Float16)signal[((size_t)b * NPTS + nidx[p]) * CH + c];
  }
  __syncthreads();

  // ---- Phase B: WMMA. Each wave owns 16 rows of C (c = 16*wv + m). ----
  const int lane = tid & 31;
  const int wv   = tid >> 5;
  const int g    = lane >> 4;     // half-wave
  const int mn   = lane & 15;     // M row (A/C) or N col (B/C)
  const int c0   = wv * 16;

  v8f acc[4] = {};                // 4 N-tiles (one per radial shell)
  #pragma unroll
  for (int kb = 0; kb < PATCH; kb += 32) {
    const _Float16* arow = &sigT[c0 + mn][0];
    const v16h a = cat8(*(const v8h*)(arow + kb + 8 * g),
                        *(const v8h*)(arow + kb + 16 + 8 * g));
    #pragma unroll
    for (int t = 0; t < 4; ++t) {
      const _Float16* brow = &kernT[16 * t + mn][0];
      const v16h bb = cat8(*(const v8h*)(brow + kb + 8 * g),
                           *(const v8h*)(brow + kb + 16 + 8 * g));
      acc[t] = __builtin_amdgcn_wmma_f32_16x16x32_f16(
          false, a, false, bb, (short)0, acc[t], false, false);
    }
  }
  // square and stage for band sums
  #pragma unroll
  for (int t = 0; t < 4; ++t)
    #pragma unroll
    for (int vr = 0; vr < 8; ++vr) {
      const float val = acc[t][vr];
      ysq[c0 + vr + 8 * g][16 * t + mn] = val * val;
    }
  __syncthreads();

  // ---- Phase C: band sums per l, sqrt, emit z[pt][c*16 + r*4 + l] f16 ----
  for (int o = tid; o < FDIM; o += 128) {
    const int c = o >> 4, r = (o >> 2) & 3, l = o & 3;
    float s = 0.0f;
    for (int n = l * l; n < (l + 1) * (l + 1); ++n) s += ysq[c][r * 16 + n];
    zbuf[(size_t)pt * FDIM + o] = (_Float16)sqrtf(fmaxf(s, 1e-4f));
  }
}

// ------- Kernel 3: out[v,i] = sum_f z[v,f] * W[i,f] + bias  (WMMA) ---------
// A = z rows (K-major), B stored transposed as wtT[i][f] (K-major).
// z chunks are staged with async global->LDS (ASYNCcnt) when available.
#define ZL_S 40   // halves; 80B rows (16B aligned)
#define WT_S 40   // halves; 80B rows (16B aligned)
__global__ __launch_bounds__(128) void out_gemm_kernel(
    const _Float16* __restrict__ zbuf, const float* __restrict__ weight,
    const float* __restrict__ biases, float* __restrict__ out) {
  __shared__ _Float16 zl[64][ZL_S];    // A chunk: [row][k]
  __shared__ _Float16 wtT[64][WT_S];   // B chunk, transposed: [i][k]
  const int tid  = threadIdx.x;
  const int v0   = blockIdx.x * 64;
  const int lane = tid & 31;
  const int wv   = tid >> 5;
  const int g    = lane >> 4;
  const int mn   = lane & 15;

  v8f acc[4] = {};
  for (int kb = 0; kb < FDIM; kb += 32) {
    __syncthreads();
    // stage z: 64 rows x 32 halves
    {
      const int row = tid >> 1, off = (tid & 1) * 16;
      const _Float16* src = zbuf + (size_t)(v0 + row) * FDIM + kb + off;
#if USE_ASYNC_LDS
      // gfx1250 async DMA path: two b128 transfers per thread, no VGPR bounce
      __builtin_amdgcn_global_load_async_to_lds_b128(
          (v4i*)src,       (v4i*)&zl[row][off],     0, 0);
      __builtin_amdgcn_global_load_async_to_lds_b128(
          (v4i*)(src + 8), (v4i*)&zl[row][off + 8], 0, 0);
#else
      *(v8h*)&zl[row][off]     = *(const v8h*)src;
      *(v8h*)&zl[row][off + 8] = *(const v8h*)(src + 8);
#endif
    }
    // stage W^T chunk (f32 -> f16): wtT[i][f], f fastest => coalesced reads
    for (int e = tid; e < 64 * 32; e += 128) {
      const int i = e >> 5, f = e & 31;
      wtT[i][f] = (_Float16)weight[(size_t)i * FDIM + kb + f];
    }
#if USE_ASYNC_LDS
    __builtin_amdgcn_s_wait_asynccnt(0);   // this wave's async copies landed
#endif
    __syncthreads();
    const _Float16* arow = &zl[wv * 16 + mn][0];
    const v16h a = cat8(*(const v8h*)(arow + 8 * g),
                        *(const v8h*)(arow + 16 + 8 * g));
    #pragma unroll
    for (int t = 0; t < 4; ++t) {
      const _Float16* brow = &wtT[16 * t + mn][0];
      const v16h bb = cat8(*(const v8h*)(brow + 8 * g),
                           *(const v8h*)(brow + 16 + 8 * g));
      acc[t] = __builtin_amdgcn_wmma_f32_16x16x32_f16(
          false, a, false, bb, (short)0, acc[t], false, false);
    }
  }
  // epilogue: bias + fp32 store
  #pragma unroll
  for (int t = 0; t < 4; ++t) {
    const int i = 16 * t + mn;
    const float bias = biases[i];
    #pragma unroll
    for (int vr = 0; vr < 8; ++vr) {
      const int row = v0 + wv * 16 + vr + 8 * g;
      out[(size_t)row * CH + i] = acc[t][vr] + bias;
    }
  }
}

// --------------------------------- launch ----------------------------------
extern "C" void kernel_launch(void* const* d_in, const int* in_sizes, int n_in,
                              void* d_out, int out_size, void* d_ws, size_t ws_size,
                              hipStream_t stream) {
  const float* xyz    = (const float*)d_in[0];
  const float* signal = (const float*)d_in[1];
  const float* weight = (const float*)d_in[2];
  const float* biases = (const float*)d_in[3];
  float* out = (float*)d_out;

  char* ws = (char*)d_ws;
  int*      knn_idx = (int*)ws;
  float*    knn_d2  = (float*)(ws + (size_t)NPOINTS * PATCH * sizeof(int));
  _Float16* zbuf    = (_Float16*)(ws + (size_t)NPOINTS * PATCH * (sizeof(int) + sizeof(float)));

  knn_kernel<<<NPOINTS, 128, 0, stream>>>(xyz, knn_idx, knn_d2);
  point_conv_kernel<<<NPOINTS, 128, 0, stream>>>(xyz, signal, knn_idx, knn_d2, zbuf);
  out_gemm_kernel<<<NPOINTS / 64, 128, 0, stream>>>(zbuf, weight, biases, out);
}
